// ClusterModel_6184752906742
// MI455X (gfx1250) — matrix operand order repaired and checked by reading, not compiled, not run on hardware
//
#include <hip/hip_runtime.h>
#include <hip/hip_bf16.h>
#include <math.h>

typedef __attribute__((ext_vector_type(16))) __bf16        v16bf;
typedef __attribute__((ext_vector_type(8)))  float         v8f;
typedef __attribute__((ext_vector_type(4)))  float         f32x4;
typedef __attribute__((ext_vector_type(4)))  unsigned int  u32x4;

#define KC 32
#define TOPK 5
#define MIN_DIST 0.1f
#define EPSV 1e-8f

// ws layout (float indices)
#define WS_OCC   0     // 32: sum Aw (raw, no eps)
#define WS_SX    32    // 96: sum Aw*x, k-major
#define WS_SXX   128   // 32: sum Aw*|x|^2
#define WS_COV   160   // 1
#define WS_PRUNE 161   // 1
#define WS_PRED  192   // P floats

__device__ __forceinline__ unsigned int pk_bf16(float a, float b) {
  unsigned int ia = __float_as_uint(a);
  ia = (ia + 0x7FFFu + ((ia >> 16) & 1u)) >> 16;
  unsigned int ib = __float_as_uint(b);
  ib = (ib + 0x7FFFu + ((ib >> 16) & 1u)) & 0xFFFF0000u;
  return ia | ib;
}
__device__ __forceinline__ unsigned short f_to_bf16(float a) {
  unsigned int ia = __float_as_uint(a);
  return (unsigned short)((ia + 0x7FFFu + ((ia >> 16) & 1u)) >> 16);
}

__global__ void k_init(float* W, int n) {
  for (int i = threadIdx.x + blockIdx.x * blockDim.x; i < n; i += blockDim.x * gridDim.x)
    W[i] = 0.f;
}

// One pass over A_logits: softmax, transposed bf16 store, cluster stats via padded-LDS
// transpose (no bpermute storms), top-5 covisibility, prune.
__global__ void __launch_bounds__(256) k_gauss_pass(
    const float* __restrict__ Alog, const float* __restrict__ x,
    const float* __restrict__ wts, float* __restrict__ W,
    unsigned short* __restrict__ At, int G) {
  __shared__ float s_aw[256][KC + 1];              // padded: stride 33 -> conflict-free
  __shared__ __align__(16) float s_mx[256][4];     // x0,x1,x2,|x|^2 per row
  __shared__ float s_part[8][KC][5];               // per-rowgroup partials
  __shared__ float s_cov, s_prune;
  const int tid = threadIdx.x;
  if (tid == 0) { s_cov = 0.f; s_prune = 0.f; }

  const int g = blockIdx.x * blockDim.x + tid;
  const bool act = (g < G);

  float v[KC];
  if (act) {
    const float* row = Alog + (size_t)g * KC;
#pragma unroll
    for (int j = 0; j < KC / 4; ++j) {
      f32x4 t = *(const f32x4*)(row + 4 * j);
      v[4*j] = t.x; v[4*j+1] = t.y; v[4*j+2] = t.z; v[4*j+3] = t.w;
    }
  } else {
#pragma unroll
    for (int k = 0; k < KC; ++k) v[k] = 0.f;
  }
  float mx = v[0];
#pragma unroll
  for (int k = 1; k < KC; ++k) mx = fmaxf(mx, v[k]);
  float s = 0.f;
#pragma unroll
  for (int k = 0; k < KC; ++k) { v[k] = expf(v[k] - mx); s += v[k]; }
  const float inv = 1.f / s;
#pragma unroll
  for (int k = 0; k < KC; ++k) v[k] *= inv;

  if (act) {
#pragma unroll
    for (int k = 0; k < KC; ++k) At[(size_t)k * G + g] = f_to_bf16(v[k]);
  }

  float w = 0.f, x0 = 0.f, x1 = 0.f, x2 = 0.f;
  if (act) {
    w  = wts[g];
    x0 = x[3 * (size_t)g]; x1 = x[3 * (size_t)g + 1]; x2 = x[3 * (size_t)g + 2];
  }
  const float xx = x0 * x0 + x1 * x1 + x2 * x2;

  // stage rows into LDS (inactive rows contribute zeros since w==0)
#pragma unroll
  for (int k = 0; k < KC; ++k) s_aw[tid][k] = v[k] * w;
  s_mx[tid][0] = x0; s_mx[tid][1] = x1; s_mx[tid][2] = x2; s_mx[tid][3] = xx;

  // covisibility: top-5 of softmax row; pair = 5*sum(v^2) - sum(v)^2
  if (act) {
    float q[KC];
#pragma unroll
    for (int k = 0; k < KC; ++k) q[k] = v[k];
    float ts = 0.f, tss = 0.f;
#pragma unroll
    for (int it = 0; it < TOPK; ++it) {
      float m = -1e30f;
#pragma unroll
      for (int k = 0; k < KC; ++k) m = fmaxf(m, q[k]);
      ts += m; tss += m * m;
      bool done = false;
#pragma unroll
      for (int k = 0; k < KC; ++k) {
        if (!done && q[k] == m) { q[k] = -1e30f; done = true; }
      }
    }
    atomicAdd(&s_cov, (float)TOPK * tss - ts * ts);
    atomicAdd(&s_prune, fabsf(w));
  }
  __syncthreads();

  // transpose-reduce: thread owns (k = tid&31, rowgroup = tid>>5), walks 32 rows
  {
    const int k = tid & (KC - 1);
    const int grp = tid >> 5;
    const int base = grp * 32;
    float a0 = 0.f, a1 = 0.f, a2 = 0.f, a3 = 0.f, a4 = 0.f;
#pragma unroll
    for (int r = 0; r < 32; ++r) {
      const float aw = s_aw[base + r][k];
      const f32x4 m = *(const f32x4*)s_mx[base + r];   // LDS broadcast across the wave
      a0 += aw;
      a1 += aw * m.x; a2 += aw * m.y; a3 += aw * m.z; a4 += aw * m.w;
    }
    s_part[grp][k][0] = a0; s_part[grp][k][1] = a1; s_part[grp][k][2] = a2;
    s_part[grp][k][3] = a3; s_part[grp][k][4] = a4;
  }
  __syncthreads();

  if (tid < KC) {
    float a0 = 0.f, a1 = 0.f, a2 = 0.f, a3 = 0.f, a4 = 0.f;
#pragma unroll
    for (int grp = 0; grp < 8; ++grp) {
      a0 += s_part[grp][tid][0]; a1 += s_part[grp][tid][1]; a2 += s_part[grp][tid][2];
      a3 += s_part[grp][tid][3]; a4 += s_part[grp][tid][4];
    }
    atomicAdd(&W[WS_OCC + tid], a0);
    atomicAdd(&W[WS_SX + 3 * tid + 0], a1);
    atomicAdd(&W[WS_SX + 3 * tid + 1], a2);
    atomicAdd(&W[WS_SX + 3 * tid + 2], a3);
    atomicAdd(&W[WS_SXX + tid], a4);
  }
  if (tid == 0) { atomicAdd(&W[WS_COV], s_cov); atomicAdd(&W[WS_PRUNE], s_prune); }
}

// ---- WMMA GEMM: pred[p] = sum_n (R @ A_bf16)[p,n], software-pipelined ----
struct TileRegs {
  f32x4 r0, r1, r2, r3;
  u32x4 b00, b01, b10, b11;
};

__device__ __forceinline__ void tile_load(TileRegs& T, const float* __restrict__ Rrow,
                                          const unsigned short* __restrict__ B0p,
                                          const unsigned short* __restrict__ B1p,
                                          int g0, int half) {
  const float* pa = Rrow + g0 + 8 * half;
  T.r0 = *(const f32x4*)(pa);
  T.r1 = *(const f32x4*)(pa + 4);
  T.r2 = *(const f32x4*)(pa + 16);
  T.r3 = *(const f32x4*)(pa + 20);
  const unsigned short* pb0 = B0p + g0 + 8 * half;
  const unsigned short* pb1 = B1p + g0 + 8 * half;
  T.b00 = *(const u32x4*)(pb0); T.b01 = *(const u32x4*)(pb0 + 16);
  T.b10 = *(const u32x4*)(pb1); T.b11 = *(const u32x4*)(pb1 + 16);
}

__device__ __forceinline__ void tile_mma(const TileRegs& T, v8f& c0, v8f& c1) {
  union UA { unsigned int u[8]; v16bf v; } A;
  union UB { u32x4 q[2];       v16bf v; } B0, B1;
  A.u[0] = pk_bf16(T.r0.x, T.r0.y); A.u[1] = pk_bf16(T.r0.z, T.r0.w);
  A.u[2] = pk_bf16(T.r1.x, T.r1.y); A.u[3] = pk_bf16(T.r1.z, T.r1.w);
  A.u[4] = pk_bf16(T.r2.x, T.r2.y); A.u[5] = pk_bf16(T.r2.z, T.r2.w);
  A.u[6] = pk_bf16(T.r3.x, T.r3.y); A.u[7] = pk_bf16(T.r3.z, T.r3.w);
  B0.q[0] = T.b00; B0.q[1] = T.b01;
  B1.q[0] = T.b10; B1.q[1] = T.b11;
  c0 = __builtin_amdgcn_wmma_f32_16x16x32_bf16(false, A.v, false, B0.v, (short)0, c0, false, false);
  c1 = __builtin_amdgcn_wmma_f32_16x16x32_bf16(false, A.v, false, B1.v, (short)0, c1, false, false);
}

__global__ void __launch_bounds__(256) k_wmma_pred(
    const float* __restrict__ R, const unsigned short* __restrict__ At,
    float* __restrict__ pred, int G) {
  const int lane = threadIdx.x & 31;
  const int wave = threadIdx.x >> 5;
  const int m    = lane & 15;
  const int half = lane >> 4;
  const int p0   = blockIdx.x * 16;

  const float* Rrow = R + (size_t)(p0 + m) * (size_t)G;
  const unsigned short* B0p = At + (size_t)m * (size_t)G;        // clusters 0..15
  const unsigned short* B1p = At + (size_t)(16 + m) * (size_t)G; // clusters 16..31

  const int wy = blockIdx.y * 8 + wave;
  const int wystride = gridDim.y * 8;
  const int nsteps = G >> 5;

  v8f c0 = {}; v8f c1 = {};
  TileRegs cur, nxt;

  int t = wy;
  bool have = (t < nsteps);
  if (have) tile_load(cur, Rrow, B0p, B1p, t << 5, half);
  while (have) {                                 // wave-uniform loop: EXEC stays full
    const int t2 = t + wystride;
    const bool have2 = (t2 < nsteps);
    if (have2) {
      tile_load(nxt, Rrow, B0p, B1p, t2 << 5, half);       // overlap with MMA below
      if (t2 + wystride < nsteps)
        __builtin_prefetch(Rrow + ((size_t)(t2 + wystride) << 5), 0, 1);
    }
    tile_mma(cur, c0, c1);
    cur = nxt;
    t = t2; have = have2;
  }

  // D layout: VGPR r -> M = r + 8*half, N = lane&15. Sum N across the 16-lane group.
#pragma unroll
  for (int r = 0; r < 8; ++r) {
    float v = c0[r] + c1[r];
#pragma unroll
    for (int mm = 8; mm >= 1; mm >>= 1) v += __shfl_xor(v, mm);
    if (m == 0) atomicAdd(&pred[p0 + 8 * half + r], v);
  }
}

__device__ __forceinline__ float block_reduce(float v, float* sred) {
  const int tid = threadIdx.x;
  sred[tid] = v; __syncthreads();
  for (int s = 128; s > 0; s >>= 1) {
    if (tid < s) sred[tid] += sred[tid + s];
    __syncthreads();
  }
  float r = sred[0]; __syncthreads();
  return r;
}

__global__ void k_finalize(const float* __restrict__ W, const int* __restrict__ imask,
                           float* __restrict__ out, int G, int P) {
  __shared__ float sc[KC][3];
  __shared__ float sred[256];
  const int tid = threadIdx.x;

  float disp_local = 0.f;
  if (tid < KC) {
    const float s1 = W[WS_OCC + tid];
    const float occ = s1 + EPSV;
    const float sx0 = W[WS_SX + 3*tid], sx1 = W[WS_SX + 3*tid + 1], sx2 = W[WS_SX + 3*tid + 2];
    const float c0 = sx0 / occ, c1 = sx1 / occ, c2 = sx2 / occ;
    sc[tid][0] = c0; sc[tid][1] = c1; sc[tid][2] = c2;
    const float sxx = W[WS_SXX + tid];
    disp_local = (sxx - 2.f * (c0*sx0 + c1*sx1 + c2*sx2) + (c0*c0 + c1*c1 + c2*c2) * s1) / occ;
  }
  const float disp = block_reduce(disp_local, sred);

  float sep_local = 0.f;
  for (int t = tid; t < KC * KC; t += 256) {
    const int i = t >> 5, j = t & 31;
    if (j > i) {
      const float dx = sc[i][0] - sc[j][0];
      const float dy = sc[i][1] - sc[j][1];
      const float dz = sc[i][2] - sc[j][2];
      const float d = sqrtf(dx*dx + dy*dy + dz*dz);
      const float pen = MIN_DIST - d;
      if (pen > 0.f) sep_local += pen * pen;
    }
  }
  const float sep = block_reduce(sep_local, sred);

  float bce_local = 0.f;
  for (int p = tid; p < P; p += 256) {
    float pc = W[WS_PRED + p];
    pc = fminf(fmaxf(pc, 0.f), 1.f);
    const float tv = (float)imask[p];
    const float lp  = fmaxf(logf(pc), -100.f);
    const float l1p = fmaxf(log1pf(-pc), -100.f);
    bce_local += tv * lp + (1.f - tv) * l1p;
  }
  const float bce = block_reduce(bce_local, sred);

  if (tid == 0) {
    const float render = -bce / (float)P;
    out[0] = render + disp + sep + W[WS_COV] / (float)G + W[WS_PRUNE];
  }
}

extern "C" void kernel_launch(void* const* d_in, const int* in_sizes, int n_in,
                              void* d_out, int out_size, void* d_ws, size_t ws_size,
                              hipStream_t stream) {
  (void)n_in; (void)out_size; (void)ws_size;
  const float* Alog = (const float*)d_in[0];
  const float* x    = (const float*)d_in[1];
  const float* wts  = (const float*)d_in[2];
  const float* R    = (const float*)d_in[3];
  const int* imask  = (const int*)d_in[4];
  float* out = (float*)d_out;

  const int G = in_sizes[2];   // 200000
  const int P = in_sizes[4];   // 2048

  float* W = (float*)d_ws;
  const size_t atOff = (((size_t)(WS_PRED + P) * 4 + 255) & ~(size_t)255);
  unsigned short* At = (unsigned short*)((char*)d_ws + atOff); // K*G bf16 = 12.8 MB

  hipLaunchKernelGGL(k_init, dim3(8), dim3(256), 0, stream, W, WS_PRED + P);
  hipLaunchKernelGGL(k_gauss_pass, dim3((G + 255) / 256), dim3(256), 0, stream,
                     Alog, x, wts, W, At, G);
  hipLaunchKernelGGL(k_wmma_pred, dim3(P / 16, 40), dim3(256), 0, stream,
                     R, At, W + WS_PRED, G);
  hipLaunchKernelGGL(k_finalize, dim3(1), dim3(256), 0, stream, W, imask, out, G, P);
}
